// Cross_Attention_26517128085816
// MI455X (gfx1250) — matrix operand order repaired and checked by reading, not compiled, not run on hardware
//
#include <hip/hip_runtime.h>
#include <stdint.h>

// Problem constants
#define CIN 512
#define DH  64
#define NS  2304          // 48*48
#define NBATCH 4
#define NT  (NS / 16)     // 144 n-tiles
#define CT  (CIN / 16)    // 32 c-tiles

typedef __attribute__((ext_vector_type(16))) __bf16 v16bf;
typedef __attribute__((ext_vector_type(8)))  float  v8f;

union Frag {
  v16bf v;
  uint4 q[2];
  unsigned u[8];
};

// f32 -> bf16 using the hardware conversion
__device__ __forceinline__ unsigned short b16(float a) {
  union { __bf16 h; unsigned short s; } u;
  u.h = (__bf16)a;
  return u.s;
}
__device__ __forceinline__ unsigned pk2(float a, float b) {
  return (unsigned)b16(a) | ((unsigned)b16(b) << 16);
}

__device__ __forceinline__ v8f wmma_bf16(v16bf a, v16bf b, v8f c) {
  // (neg_a, A, neg_b, B, c_mod, C, reuse_a, reuse_b)
  return __builtin_amdgcn_wmma_f32_16x16x32_bf16(false, a, false, b, (short)0, c,
                                                 false, false);
}

// A-fragment (16x32 bf16): lane's chunks at rowp[0..7] and rowp[8..15]
// (caller applied koff = (lane>>4)*16)
__device__ __forceinline__ Frag load_afrag_bf16(const unsigned short* __restrict__ rowp) {
  Frag f;
  f.q[0] = *(const uint4*)(rowp);
  f.q[1] = *(const uint4*)(rowp + 8);
  return f;
}

// B-fragment: 16 contiguous bf16 per lane
__device__ __forceinline__ Frag load_bfrag_bf16(const unsigned short* __restrict__ p) {
  Frag f;
  f.q[0] = *(const uint4*)(p);
  f.q[1] = *(const uint4*)(p + 8);
  return f;
}

// Workspace layout (ushort elements)
#define XT_OFF   ((size_t)0)
#define YT_OFF   ((size_t)NBATCH * NS * CIN)                 // 4,718,592
#define WBF_OFF  ((size_t)2 * NBATCH * NS * CIN)             // 9,437,184
#define WMAT     ((size_t)32768)                             // each weight = 64*512
#define PRJ_OFF  (WBF_OFF + 10 * WMAT)                       // 9,764,864
#define SEG      ((size_t)NBATCH * NS * DH)                  // 589,824
// PRJ segments: 0..5 = F1,G1,F2,G2,H1,H2 ; 6..9 = AGG11,AGG21,AGG22,AGG12

// ---------------------------------------------------------------------------
// Prep A: convert 10 weight matrices to bf16 (order: Wf1,Wg1,Wf2,Wg2,Wh1,Wh2,
// Wv11,Wv21,Wv22,Wv12), each 32768 elements.
// ---------------------------------------------------------------------------
__global__ __launch_bounds__(256) void prep_weights(
    const float* __restrict__ Wf1, const float* __restrict__ Wg1,
    const float* __restrict__ Wf2, const float* __restrict__ Wg2,
    const float* __restrict__ Wh1, const float* __restrict__ Wh2,
    const float* __restrict__ Wv11, const float* __restrict__ Wv21,
    const float* __restrict__ Wv22, const float* __restrict__ Wv12,
    unsigned short* __restrict__ dst)
{
  int tid = blockIdx.x * blockDim.x + threadIdx.x;
  if (tid >= 10 * (int)WMAT) return;
  int m = tid >> 15;
  int i = tid & 32767;
  const float* s;
  switch (m) {
    case 0: s = Wf1;  break;
    case 1: s = Wg1;  break;
    case 2: s = Wf2;  break;
    case 3: s = Wg2;  break;
    case 4: s = Wh1;  break;
    case 5: s = Wh2;  break;
    case 6: s = Wv11; break;
    case 7: s = Wv21; break;
    case 8: s = Wv22; break;
    default: s = Wv12; break;
  }
  dst[tid] = b16(s[i]);
}

// ---------------------------------------------------------------------------
// Prep B: transpose + convert x,y [b][c][n] f32 -> [b][n][c] bf16.
// 32x32 LDS tile, coalesced reads and writes. 256 threads = 32x8.
// ---------------------------------------------------------------------------
__global__ __launch_bounds__(256) void prep_xpose(
    const float* __restrict__ x, const float* __restrict__ y,
    unsigned short* __restrict__ xt, unsigned short* __restrict__ yt)
{
  __shared__ float tile[32][33];
  int tid = blockIdx.x;
  int which = tid & 1;
  tid >>= 1;
  int ctile = tid % (CIN / 32);
  int rest  = tid / (CIN / 32);
  int ntile = rest % (NS / 32);
  int b     = rest / (NS / 32);
  const float* src = which ? y : x;
  unsigned short* dst = which ? yt : xt;

  int tx = threadIdx.x & 31;
  int ty = threadIdx.x >> 5;          // 0..7
  int n0 = ntile * 32, c0 = ctile * 32;

#pragma unroll
  for (int k = 0; k < 4; k++) {
    int cl = ty + k * 8;
    tile[cl][tx] = src[((size_t)b * CIN + c0 + cl) * NS + n0 + tx];
  }
  __syncthreads();
#pragma unroll
  for (int k = 0; k < 4; k++) {
    int nl = ty + k * 8;
    dst[((size_t)b * NS + n0 + nl) * CIN + c0 + tx] = b16(tile[tx][nl]);
  }
}

// ---------------------------------------------------------------------------
// Kernel 1: six input projections.  out[d][n] = sum_c W[d][c] * z[n][c] + bias
// f1,g1,f2,g2 -> bf16 [b][n][d] ; h1,h2 -> bf16 [b][d][n]
// One wave: 64(d) x 16(n) tile.  Software-pipelined: stage k+1 fragments are
// loaded before stage k's WMMAs so loads stay >=1 stage in flight.
// ---------------------------------------------------------------------------
__global__ __launch_bounds__(256) void proj_kernel(
    const float* __restrict__ bf1, const float* __restrict__ bg1,
    const float* __restrict__ bh1, const float* __restrict__ bf2,
    const float* __restrict__ bg2, const float* __restrict__ bh2,
    unsigned short* __restrict__ ws)
{
  int wid  = (blockIdx.x * blockDim.x + threadIdx.x) >> 5;
  int lane = threadIdx.x & 31;
  if (wid >= 6 * NBATCH * NT) return;
  int nt = wid % NT;
  int b  = (wid / NT) % NBATCH;
  int p  = wid / (NT * NBATCH);

  const unsigned short* W;
  const unsigned short* srcT;
  const float* bias;
  unsigned short* dst;
  int dn;
  switch (p) {
    case 0:  W = ws + WBF_OFF + 0*WMAT; bias = bf1; srcT = ws + XT_OFF; dst = ws + PRJ_OFF + 0*SEG; dn = 0; break;
    case 1:  W = ws + WBF_OFF + 1*WMAT; bias = bg1; srcT = ws + XT_OFF; dst = ws + PRJ_OFF + 1*SEG; dn = 0; break;
    case 2:  W = ws + WBF_OFF + 2*WMAT; bias = bf2; srcT = ws + YT_OFF; dst = ws + PRJ_OFF + 2*SEG; dn = 0; break;
    case 3:  W = ws + WBF_OFF + 3*WMAT; bias = bg2; srcT = ws + YT_OFF; dst = ws + PRJ_OFF + 3*SEG; dn = 0; break;
    case 4:  W = ws + WBF_OFF + 4*WMAT; bias = bh1; srcT = ws + XT_OFF; dst = ws + PRJ_OFF + 4*SEG; dn = 1; break;
    default: W = ws + WBF_OFF + 5*WMAT; bias = bh2; srcT = ws + XT_OFF; dst = ws + PRJ_OFF + 5*SEG; dn = 1; break;
  }

  int half = lane >> 4;
  int l15  = lane & 15;
  int col  = nt * 16 + l15;      // n index
  int koff = half * 16;

  const unsigned short* bbase = srcT + ((size_t)b * NS + col) * CIN + half * 16;
  const unsigned short* abase = W + (size_t)l15 * CIN + koff;

  v8f acc[4] = {};

  // prologue: stage-0 fragments
  Frag bcur = load_bfrag_bf16(bbase);
  Frag a0 = load_afrag_bf16(abase + 0 * 16 * CIN);
  Frag a1 = load_afrag_bf16(abase + 1 * 16 * CIN);
  Frag a2 = load_afrag_bf16(abase + 2 * 16 * CIN);
  Frag a3 = load_afrag_bf16(abase + 3 * 16 * CIN);

#pragma unroll
  for (int c0 = 0; c0 < CIN; c0 += 32) {
    Frag bnx, n0, n1, n2, n3;
    if (c0 + 32 < CIN) {                 // issue next stage's loads first
      bnx = load_bfrag_bf16(bbase + c0 + 32);
      n0  = load_afrag_bf16(abase + c0 + 32 + 0 * 16 * CIN);
      n1  = load_afrag_bf16(abase + c0 + 32 + 1 * 16 * CIN);
      n2  = load_afrag_bf16(abase + c0 + 32 + 2 * 16 * CIN);
      n3  = load_afrag_bf16(abase + c0 + 32 + 3 * 16 * CIN);
    }
    acc[0] = wmma_bf16(a0.v, bcur.v, acc[0]);
    acc[1] = wmma_bf16(a1.v, bcur.v, acc[1]);
    acc[2] = wmma_bf16(a2.v, bcur.v, acc[2]);
    acc[3] = wmma_bf16(a3.v, bcur.v, acc[3]);
    bcur = bnx; a0 = n0; a1 = n1; a2 = n2; a3 = n3;
  }

  if (!dn) {
    // [b][n][d]: this lane owns 8 consecutive d per tile -> one 16B store
#pragma unroll
    for (int t = 0; t < 4; t++) {
      int dbase = t * 16 + half * 8;
      unsigned u0 = pk2(acc[t][0] + bias[dbase + 0], acc[t][1] + bias[dbase + 1]);
      unsigned u1 = pk2(acc[t][2] + bias[dbase + 2], acc[t][3] + bias[dbase + 3]);
      unsigned u2 = pk2(acc[t][4] + bias[dbase + 4], acc[t][5] + bias[dbase + 5]);
      unsigned u3 = pk2(acc[t][6] + bias[dbase + 6], acc[t][7] + bias[dbase + 7]);
      uint4 st = make_uint4(u0, u1, u2, u3);
      *(uint4*)(dst + ((size_t)b * NS + col) * DH + dbase) = st;
    }
  } else {
    // [b][d][n]: coalesced 2B stores across lanes (consecutive n)
#pragma unroll
    for (int t = 0; t < 4; t++) {
#pragma unroll
      for (int r = 0; r < 8; r++) {
        int d = t * 16 + half * 8 + r;
        dst[((size_t)b * DH + d) * NS + col] = b16(acc[t][r] + bias[d]);
      }
    }
  }
}

// ---------------------------------------------------------------------------
// Kernel 2: fused flash attention.  One wave = (pair, batch, 16-row i-tile).
// All G and H fragments for a 32-j step are issued up front so the S-WMMAs,
// softmax VALU and LDS transpose overlap the H loads.
// ---------------------------------------------------------------------------
__global__ __launch_bounds__(128) void attn_kernel(unsigned short* __restrict__ ws)
{
  __shared__ __align__(16) unsigned short pbuf[4][16][40];  // per-wave P tile, padded

  int wid  = (blockIdx.x * blockDim.x + threadIdx.x) >> 5;
  int lane = threadIdx.x & 31;
  if (wid >= 4 * NBATCH * NT) return;
  int it = wid % NT;
  int b  = (wid / NT) % NBATCH;
  int p  = wid / (NT * NBATCH);
  unsigned short (*pb)[40] = pbuf[(threadIdx.x >> 5) & 3];

  unsigned short* prj = ws + PRJ_OFF;
  const unsigned short *F, *G, *H;
  unsigned short* AG;
  switch (p) {
    case 0:  F = prj + 0*SEG; G = prj + 1*SEG; H = prj + 4*SEG; AG = prj + 6*SEG; break; // 11
    case 1:  F = prj + 2*SEG; G = prj + 1*SEG; H = prj + 4*SEG; AG = prj + 7*SEG; break; // 21
    case 2:  F = prj + 2*SEG; G = prj + 3*SEG; H = prj + 5*SEG; AG = prj + 8*SEG; break; // 22
    default: F = prj + 0*SEG; G = prj + 3*SEG; H = prj + 5*SEG; AG = prj + 9*SEG; break; // 12
  }

  int half = lane >> 4;
  int l15  = lane & 15;
  int i0   = it * 16;
  int koff = half * 16;

  // Q A-fragments (16 rows x K=64), resident for the whole j loop
  const unsigned short* fr = F + ((size_t)b * NS + i0 + l15) * DH;
  Frag qa0 = load_afrag_bf16(fr + koff);
  Frag qa1 = load_afrag_bf16(fr + 32 + koff);

  float M[8], l[8];
  v8f O[4] = {};
#pragma unroll
  for (int r = 0; r < 8; r++) { M[r] = -1e30f; l[r] = 0.f; }

  for (int jt = 0; jt < NS; jt += 32) {
    // --- issue ALL loads for this 32-j step up front ---
    Frag kb[4], hb[4];
#pragma unroll
    for (int sub = 0; sub < 2; sub++) {
      const unsigned short* gr =
          G + ((size_t)b * NS + jt + sub * 16 + l15) * DH + half * 16;
      kb[2 * sub + 0] = load_bfrag_bf16(gr);
      kb[2 * sub + 1] = load_bfrag_bf16(gr + 32);
    }
#pragma unroll
    for (int t = 0; t < 4; t++) {
      const unsigned short* hr =
          H + ((size_t)b * DH + t * 16 + l15) * NS + jt + half * 16;
      hb[t] = load_bfrag_bf16(hr);
    }

    // --- S tiles for 32 j's (two 16-wide subtiles) ---
    v8f s[2];
    {
      v8f z0 = {}, z1 = {};
      s[0] = wmma_bf16(qa1.v, kb[1].v, wmma_bf16(qa0.v, kb[0].v, z0));
      s[1] = wmma_bf16(qa1.v, kb[3].v, wmma_bf16(qa0.v, kb[2].v, z1));
    }

    // --- per-row max over these 32 cols (xor reduce stays within lane half) ---
    float rm[8];
#pragma unroll
    for (int r = 0; r < 8; r++) rm[r] = fmaxf(s[0][r], s[1][r]);
#pragma unroll
    for (int d2 = 1; d2 < 16; d2 <<= 1) {
#pragma unroll
      for (int r = 0; r < 8; r++) rm[r] = fmaxf(rm[r], __shfl_xor(rm[r], d2, 32));
    }

    float p0[8], p1[8], rs[8];
#pragma unroll
    for (int r = 0; r < 8; r++) {
      float nm = fmaxf(M[r], rm[r]);
      float sc = __expf(M[r] - nm);
      M[r] = nm;
      p0[r] = __expf(s[0][r] - nm);
      p1[r] = __expf(s[1][r] - nm);
      rs[r] = p0[r] + p1[r];
      l[r] *= sc;
      O[0][r] *= sc; O[1][r] *= sc; O[2][r] *= sc; O[3][r] *= sc;
    }
#pragma unroll
    for (int d2 = 1; d2 < 16; d2 <<= 1) {
#pragma unroll
      for (int r = 0; r < 8; r++) rs[r] += __shfl_xor(rs[r], d2, 32);
    }
#pragma unroll
    for (int r = 0; r < 8; r++) l[r] += rs[r];

    // --- transpose P (C/D layout -> A layout) via per-wave LDS tile ---
#pragma unroll
    for (int r = 0; r < 8; r++) {
      pb[half * 8 + r][l15]      = b16(p0[r]);
      pb[half * 8 + r][16 + l15] = b16(p1[r]);
    }
    Frag pf;  // A-fragment of P (16 x 32), rows = i, K = j
    pf.q[0] = *(const uint4*)&pb[l15][koff];
    pf.q[1] = *(const uint4*)&pb[l15][koff + 8];

    // --- O += P . h  (hb fragments already resident) ---
#pragma unroll
    for (int t = 0; t < 4; t++) {
      O[t] = wmma_bf16(pf.v, hb[t].v, O[t]);
    }
  }

  // normalize and store agg[b][n][d] as bf16
#pragma unroll
  for (int r = 0; r < 8; r++) {
    float inv = 1.0f / l[r];
    int i = i0 + half * 8 + r;
#pragma unroll
    for (int t = 0; t < 4; t++) {
      AG[((size_t)b * NS + i) * DH + t * 16 + l15] = b16(O[t][r] * inv);
    }
  }
}

// ---------------------------------------------------------------------------
// Kernel 3: output projections + scaled residual.
// x_out = x + alpha*(Wv11.agg11+bv11) + beta*(Wv21.agg21+bv21)
// y_out = y + gamma*(Wv12.agg12+bv12) + sigma*(Wv22.agg22+bv22)
// One wave: 16(c) x 16(n) tile, K=64; all 8 fragments loaded up front.
// ---------------------------------------------------------------------------
__global__ __launch_bounds__(256) void outproj_kernel(
    const float* __restrict__ x, const float* __restrict__ y,
    const float* __restrict__ bv11, const float* __restrict__ bv12,
    const float* __restrict__ bv21, const float* __restrict__ bv22,
    const float* __restrict__ alpha, const float* __restrict__ beta,
    const float* __restrict__ gamma_, const float* __restrict__ sigma,
    const unsigned short* __restrict__ ws,
    float* __restrict__ out)
{
  int wid  = (blockIdx.x * blockDim.x + threadIdx.x) >> 5;
  int lane = threadIdx.x & 31;
  if (wid >= 2 * NBATCH * CT * NT) return;
  int nt = wid % NT;
  int ct = (wid / NT) % CT;
  int b  = (wid / (NT * CT)) % NBATCH;
  int o  = wid / (NT * CT * NBATCH);   // 0 = x-side, 1 = y-side

  const unsigned short* prj = ws + PRJ_OFF;
  const unsigned short* wbf = ws + WBF_OFF;
  const unsigned short *WA, *WB, *gA, *gB;
  const float *biasA, *biasB, *z;
  float sA, sB;
  float* dst;
  if (o == 0) {
    WA = wbf + 6*WMAT; biasA = bv11; gA = prj + 6*SEG; sA = *alpha;   // Wv11.agg11
    WB = wbf + 7*WMAT; biasB = bv21; gB = prj + 7*SEG; sB = *beta;    // Wv21.agg21
    z = x; dst = out;
  } else {
    WA = wbf + 9*WMAT; biasA = bv12; gA = prj + 9*SEG; sA = *gamma_;  // Wv12.agg12
    WB = wbf + 8*WMAT; biasB = bv22; gB = prj + 8*SEG; sB = *sigma;   // Wv22.agg22
    z = y; dst = out + (size_t)NBATCH * CIN * NS;
  }

  int half = lane >> 4;
  int l15  = lane & 15;
  int c0   = ct * 16;
  int n    = nt * 16 + l15;
  int koff = half * 16;

  // all fragments first, then the 4 WMMAs
  Frag aA0 = load_afrag_bf16(WA + (size_t)(c0 + l15) * DH + 0  + koff);
  Frag aA1 = load_afrag_bf16(WA + (size_t)(c0 + l15) * DH + 32 + koff);
  Frag aB0 = load_afrag_bf16(WB + (size_t)(c0 + l15) * DH + 0  + koff);
  Frag aB1 = load_afrag_bf16(WB + (size_t)(c0 + l15) * DH + 32 + koff);
  Frag bA0 = load_bfrag_bf16(gA + ((size_t)b * NS + n) * DH + 0  + half * 16);
  Frag bA1 = load_bfrag_bf16(gA + ((size_t)b * NS + n) * DH + 32 + half * 16);
  Frag bB0 = load_bfrag_bf16(gB + ((size_t)b * NS + n) * DH + 0  + half * 16);
  Frag bB1 = load_bfrag_bf16(gB + ((size_t)b * NS + n) * DH + 32 + half * 16);

  v8f sa = {}, sb = {};
  sa = wmma_bf16(aA0.v, bA0.v, sa);
  sa = wmma_bf16(aA1.v, bA1.v, sa);
  sb = wmma_bf16(aB0.v, bB0.v, sb);
  sb = wmma_bf16(aB1.v, bB1.v, sb);

#pragma unroll
  for (int r = 0; r < 8; r++) {
    int c = c0 + half * 8 + r;
    size_t off = ((size_t)b * CIN + c) * NS + n;
    dst[off] = z[off] + sA * (sa[r] + biasA[c]) + sB * (sb[r] + biasB[c]);
  }
}

// ---------------------------------------------------------------------------
extern "C" void kernel_launch(void* const* d_in, const int* in_sizes, int n_in,
                              void* d_out, int out_size, void* d_ws, size_t ws_size,
                              hipStream_t stream)
{
  const float* x    = (const float*)d_in[0];
  const float* y    = (const float*)d_in[1];
  const float* Wf1  = (const float*)d_in[2];  const float* bf1 = (const float*)d_in[3];
  const float* Wg1  = (const float*)d_in[4];  const float* bg1 = (const float*)d_in[5];
  const float* Wh1  = (const float*)d_in[6];  const float* bh1 = (const float*)d_in[7];
  const float* Wf2  = (const float*)d_in[8];  const float* bf2 = (const float*)d_in[9];
  const float* Wg2  = (const float*)d_in[10]; const float* bg2 = (const float*)d_in[11];
  const float* Wh2  = (const float*)d_in[12]; const float* bh2 = (const float*)d_in[13];
  const float* Wv11 = (const float*)d_in[14]; const float* bv11 = (const float*)d_in[15];
  const float* Wv12 = (const float*)d_in[16]; const float* bv12 = (const float*)d_in[17];
  const float* Wv21 = (const float*)d_in[18]; const float* bv21 = (const float*)d_in[19];
  const float* Wv22 = (const float*)d_in[20]; const float* bv22 = (const float*)d_in[21];
  const float* alpha  = (const float*)d_in[22];
  const float* beta   = (const float*)d_in[23];
  const float* gamma_ = (const float*)d_in[24];
  const float* sigma  = (const float*)d_in[25];

  unsigned short* ws = (unsigned short*)d_ws;
  float* out = (float*)d_out;

  {
    int total = 10 * (int)WMAT;                // 327,680
    prep_weights<<<(total + 255) / 256, 256, 0, stream>>>(
        Wf1, Wg1, Wf2, Wg2, Wh1, Wh2, Wv11, Wv21, Wv22, Wv12, ws + WBF_OFF);
  }
  {
    int blocks = 2 * NBATCH * (NS / 32) * (CIN / 32);  // 9216
    prep_xpose<<<blocks, 256, 0, stream>>>(x, y, ws + XT_OFF, ws + YT_OFF);
  }
  {
    int waves = 6 * NBATCH * NT;               // 3456
    proj_kernel<<<(waves * 32) / 256, 256, 0, stream>>>(
        bf1, bg1, bh1, bf2, bg2, bh2, ws);
  }
  {
    int waves = 4 * NBATCH * NT;               // 2304
    attn_kernel<<<(waves * 32) / 128, 128, 0, stream>>>(ws);
  }
  {
    int waves = 2 * NBATCH * CT * NT;          // 36864
    outproj_kernel<<<(waves * 32) / 256, 256, 0, stream>>>(
        x, y, bv11, bv12, bv21, bv22, alpha, beta, gamma_, sigma, ws, out);
  }
}
